// GLN_52261162058224
// MI455X (gfx1250) — matrix operand order repaired and checked by reading, not compiled, not run on hardware
//
#include <hip/hip_runtime.h>
#include <hip/hip_bf16.h>

// ---------------- problem constants ----------------
#define NN    20000   // nodes
#define NG    64      // graphs
#define NLAB  4       // labels
#define NE    320000  // edges per label
#define FIN   128     // input features
#define NH    512     // hidden
#define NLAY  4       // layers

typedef __attribute__((ext_vector_type(16))) __bf16 v16bf;
typedef __attribute__((ext_vector_type(8)))  float  v8f;

union FragBF16 {
    uint4 q[2];   // 32 bytes
    v16bf v;
};

__device__ __forceinline__ unsigned short f2bf(float f) {
    unsigned int u = __float_as_uint(f);
    unsigned int r = u + 0x7FFFu + ((u >> 16) & 1u);   // round-to-nearest-even
    return (unsigned short)(r >> 16);
}

// ---------------- weight transpose + cast: W[K][N] f32 -> Wt[N][K] bf16 ----------------
__global__ void k_transpose_cast(const float* __restrict__ W, unsigned short* __restrict__ Wt,
                                 int K, int N) {
    int t = blockIdx.x * 256 + threadIdx.x;
    if (t >= K * N) return;
    int n = t / K;
    int k = t - n * K;
    Wt[t] = f2bf(W[(size_t)k * N + n]);
}

// ---------------- elementwise f32 -> bf16 (optional relu), vectorized x4 ----------------
__global__ void k_cast_bf16(const float* __restrict__ in, unsigned short* __restrict__ out,
                            int n4, int relu) {
    int t = blockIdx.x * 256 + threadIdx.x;
    if (t >= n4) return;
    float4 v = ((const float4*)in)[t];
    if (relu) {
        v.x = fmaxf(v.x, 0.f); v.y = fmaxf(v.y, 0.f);
        v.z = fmaxf(v.z, 0.f); v.w = fmaxf(v.w, 0.f);
    }
    ushort4 o;
    o.x = f2bf(v.x); o.y = f2bf(v.y); o.z = f2bf(v.z); o.w = f2bf(v.w);
    ((ushort4*)out)[t] = o;
}

__global__ void k_zero_f32(float* __restrict__ p, int n) {
    int t = blockIdx.x * 256 + threadIdx.x;
    if (t < n) p[t] = 0.f;
}

// ---------------- WMMA bf16 GEMM: C[M][N] = A[M][K] * B[K][N] (+bias) ----------------
// A: bf16 row-major [M][K];  Bt: bf16 [N][K] (B column-major);  C: f32 row-major.
// 8 waves per block; each wave computes a 32x64 tile (2 M-subtiles x 4 N-subtiles,
// 8 f32 accumulators). K loop is software-pipelined: fragments for step k+32 are
// fetched before the WMMAs of step k are issued, so global loads overlap the
// matrix pipe instead of serializing on s_wait_loadcnt 0.
__global__ __launch_bounds__(256) void k_wmma_gemm(
    const unsigned short* __restrict__ A,
    const unsigned short* __restrict__ Bt,
    const float* __restrict__ bias,      // may be nullptr
    float* __restrict__ C,
    int M, int K, int N)
{
    const int wave = threadIdx.x >> 5;
    const int lane = threadIdx.x & 31;
    const int m0 = (blockIdx.x * 8 + wave) * 32;       // 32 rows per wave
    if (m0 >= M) return;                               // wave-uniform exit

    const int n0   = blockIdx.y * 64;
    const int l16  = lane & 15;
    const int half = lane >> 4;                        // 0 or 1

    // A fragment 16x32 (bf16): half 0 -> K slots {kb..kb+7, kb+16..kb+23}
    //                          half 1 -> K slots {kb+8..kb+15, kb+24..kb+31}
    const unsigned short* arow0 = A + (size_t)(m0 + l16) * K;
    const unsigned short* arow1 = arow0 + (size_t)16 * K;
    // B fragment 32x16 (bf16): lane col = n0+t*16+l16, K slots = {kb+half*8..+7, kb+16+half*8..+7}
    const unsigned short* bbase = Bt + (size_t)(n0 + l16) * K + half * 8;

    v8f acc[8];
#pragma unroll
    for (int i = 0; i < 8; ++i) acc[i] = (v8f){};

    FragBF16 aC[2], bC[4], aN[2], bN[4];

    auto loadA = [&](FragBF16* f, int kb) {
        f[0].q[0] = *(const uint4*)(arow0 + kb +      half * 8);
        f[0].q[1] = *(const uint4*)(arow0 + kb + 16 + half * 8);
        f[1].q[0] = *(const uint4*)(arow1 + kb +      half * 8);
        f[1].q[1] = *(const uint4*)(arow1 + kb + 16 + half * 8);
    };
    auto loadB = [&](FragBF16* f, int kb) {
#pragma unroll
        for (int t = 0; t < 4; ++t) {
            const unsigned short* p = bbase + (size_t)t * 16 * K + kb;
            f[t].q[0] = *(const uint4*)(p);
            f[t].q[1] = *(const uint4*)(p + 16);
        }
    };
    auto mma = [&](const FragBF16* a, const FragBF16* b) {
#pragma unroll
        for (int mi = 0; mi < 2; ++mi)
#pragma unroll
            for (int t = 0; t < 4; ++t)
                acc[mi * 4 + t] = __builtin_amdgcn_wmma_f32_16x16x32_bf16(
                    false, a[mi].v, false, b[t].v, (short)0, acc[mi * 4 + t], false, false);
    };

    loadA(aC, 0);
    loadB(bC, 0);
    for (int kb = 32; kb < K; kb += 32) {
        loadA(aN, kb);          // prefetch next k-step
        loadB(bN, kb);
        mma(aC, bC);            // compute on previously loaded fragments
#pragma unroll
        for (int i = 0; i < 2; ++i) aC[i] = aN[i];
#pragma unroll
        for (int i = 0; i < 4; ++i) bC[i] = bN[i];
    }
    mma(aC, bC);

    // C layout: lane col = l16 (within subtile); rows = half*8 + i (+ mi*16)
    const int rb = m0 + half * 8;
#pragma unroll
    for (int mi = 0; mi < 2; ++mi) {
#pragma unroll
        for (int t = 0; t < 4; ++t) {
            const int col = n0 + t * 16 + l16;
            const float bv = bias ? bias[col] : 0.f;
#pragma unroll
            for (int i = 0; i < 8; ++i) {
                C[(size_t)(rb + mi * 16 + i) * N + col] = acc[mi * 4 + t][i] + bv;
            }
        }
    }
}

// ---------------- edge scatter: out[dst[e]] += hr[src[e]] (8 floats / thread) ----------------
__global__ void k_scatter_add(const int* __restrict__ src, const int* __restrict__ dst,
                              const float* __restrict__ hr, float* __restrict__ out) {
    int t = blockIdx.x * 256 + threadIdx.x;
    if (t >= NE * (NH / 8)) return;
    int e = t >> 6;                 // NH/8 = 64 chunks per edge
    int f = (t & 63) << 3;
    int s = src[e];
    int d = dst[e];
    const float4 v0 = *(const float4*)(hr + (size_t)s * NH + f);
    const float4 v1 = *(const float4*)(hr + (size_t)s * NH + f + 4);
    float* o = out + (size_t)d * NH + f;
    atomicAdd(o + 0, v0.x); atomicAdd(o + 1, v0.y);
    atomicAdd(o + 2, v0.z); atomicAdd(o + 3, v0.w);
    atomicAdd(o + 4, v1.x); atomicAdd(o + 5, v1.y);
    atomicAdd(o + 6, v1.z); atomicAdd(o + 7, v1.w);
}

// ---------------- pooling: sums[batch[n]] += h[n]; cnt[batch[n]] += 1 ----------------
__global__ void k_pool_sum(const float* __restrict__ h, const int* __restrict__ batch,
                           float* __restrict__ sums) {
    int t = blockIdx.x * 256 + threadIdx.x;
    if (t >= NN * (NH / 4)) return;
    int n = t >> 7;
    int f = (t & 127) << 2;
    int g = batch[n];
    float4 v = *(const float4*)(h + (size_t)n * NH + f);
    float* o = sums + (size_t)g * NH + f;
    atomicAdd(o + 0, v.x);
    atomicAdd(o + 1, v.y);
    atomicAdd(o + 2, v.z);
    atomicAdd(o + 3, v.w);
}

__global__ void k_pool_cnt(const int* __restrict__ batch, float* __restrict__ cnt) {
    int n = blockIdx.x * 256 + threadIdx.x;
    if (n < NN) atomicAdd(&cnt[batch[n]], 1.0f);
}

// ---------------- final MLP: out[g] = relu(pooled@w1+b1) @ w2 + b2 ----------------
__global__ __launch_bounds__(256) void k_mlp(const float* __restrict__ sums,
                                             const float* __restrict__ cnt,
                                             const float* __restrict__ w1,
                                             const float* __restrict__ b1,
                                             const float* __restrict__ w2,
                                             const float* __restrict__ b2,
                                             float* __restrict__ out) {
    __shared__ float pooled[NH];
    __shared__ float red[256];
    const int g = blockIdx.x;
    const int tid = threadIdx.x;
    const float c = fmaxf(cnt[g], 1.0f);
    for (int j = tid; j < NH; j += 256) pooled[j] = sums[(size_t)g * NH + j] / c;
    __syncthreads();
    float part = 0.f;
    for (int j = tid; j < NH; j += 256) {
        float acc = b1[j];
        for (int k = 0; k < NH; ++k) acc += pooled[k] * w1[(size_t)k * NH + j];
        acc = fmaxf(acc, 0.f);
        part += acc * w2[j];
    }
    red[tid] = part;
    __syncthreads();
    for (int s = 128; s > 0; s >>= 1) {
        if (tid < s) red[tid] += red[tid + s];
        __syncthreads();
    }
    if (tid == 0) out[g] = red[0] + b2[0];
}

// ============================================================================
extern "C" void kernel_launch(void* const* d_in, const int* in_sizes, int n_in,
                              void* d_out, int out_size, void* d_ws, size_t ws_size,
                              hipStream_t stream) {
    const float* x      = (const float*)d_in[0];
    const int*   ei     = (const int*)  d_in[1];   // [NLAB][2][NE]
    const int*   batch  = (const int*)  d_in[2];
    const float* emb_w  = (const float*)d_in[3];   // [FIN][NH]
    const float* emb_b  = (const float*)d_in[4];
    const float* conv_w = (const float*)d_in[5];   // [NLAY][NLAB][NH][NH]
    const float* root_w = (const float*)d_in[6];   // [NLAY][NH][NH]
    const float* root_b = (const float*)d_in[7];   // [NLAY][NH]
    const float* w1     = (const float*)d_in[8];
    const float* b1     = (const float*)d_in[9];
    const float* w2     = (const float*)d_in[10];
    const float* b2     = (const float*)d_in[11];
    float* out = (float*)d_out;

    // ---- workspace carve-out (256B aligned slices) ----
    char* base = (char*)d_ws;
    size_t off = 0;
    auto carve = [&](size_t bytes) -> void* {
        void* p = base + off;
        off = (off + bytes + 255) & ~(size_t)255;
        return p;
    };
    unsigned short* h_bf   = (unsigned short*)carve((size_t)NN * NH * 2);
    float*          hnext  = (float*)carve((size_t)NN * NH * 4);
    float*          hr     = (float*)carve((size_t)NN * NH * 4);
    unsigned short* x_bf   = (unsigned short*)carve((size_t)NN * FIN * 2);
    unsigned short* embT   = (unsigned short*)carve((size_t)FIN * NH * 2);          // [NH][FIN]
    unsigned short* rootT  = (unsigned short*)carve((size_t)NLAY * NH * NH * 2);    // [l][NH][NH]
    unsigned short* convT  = (unsigned short*)carve((size_t)NLAY * NLAB * NH * NH * 2);
    float*          sums   = (float*)carve((size_t)(NG * NH + NG) * 4);             // sums + cnt
    float*          cnt    = sums + (size_t)NG * NH;
    (void)ws_size; (void)in_sizes; (void)n_in; (void)out_size;

    const dim3 B(256);

    // ---- weight conversion (f32 [K][N] -> bf16 [N][K]) ----
    k_transpose_cast<<<dim3((FIN * NH + 255) / 256), B, 0, stream>>>(emb_w, embT, FIN, NH);
    for (int l = 0; l < NLAY; ++l)
        k_transpose_cast<<<dim3((NH * NH + 255) / 256), B, 0, stream>>>(
            root_w + (size_t)l * NH * NH, rootT + (size_t)l * NH * NH, NH, NH);
    for (int lr = 0; lr < NLAY * NLAB; ++lr)
        k_transpose_cast<<<dim3((NH * NH + 255) / 256), B, 0, stream>>>(
            conv_w + (size_t)lr * NH * NH, convT + (size_t)lr * NH * NH, NH, NH);

    // ---- embedding: h = x @ emb_w + emb_b ----
    k_cast_bf16<<<dim3((NN * FIN / 4 + 255) / 256), B, 0, stream>>>(x, x_bf, NN * FIN / 4, 0);
    const int mTiles32 = (NN + 31) / 32;               // 625
    const dim3 G((mTiles32 + 7) / 8, NH / 64);         // (79, 8)
    k_wmma_gemm<<<G, B, 0, stream>>>(x_bf, embT, emb_b, hnext, NN, FIN, NH);
    k_cast_bf16<<<dim3((NN * NH / 4 + 255) / 256), B, 0, stream>>>(hnext, h_bf, NN * NH / 4, 0);

    // ---- layers ----
    const dim3 Gs((NE * (NH / 8) + 255) / 256);        // scatter grid
    for (int l = 0; l < NLAY; ++l) {
        // root transform into hnext (overwrites, so acts as the accumulator base)
        k_wmma_gemm<<<G, B, 0, stream>>>(h_bf, rootT + (size_t)l * NH * NH,
                                         root_b + (size_t)l * NH, hnext, NN, NH, NH);
        for (int r = 0; r < NLAB; ++r) {
            // hr = h @ conv_w[l][r]
            k_wmma_gemm<<<G, B, 0, stream>>>(h_bf, convT + (size_t)(l * NLAB + r) * NH * NH,
                                             nullptr, hr, NN, NH, NH);
            // hnext[dst] += hr[src]
            const int* src = ei + ((size_t)r * 2 + 0) * NE;
            const int* dst = ei + ((size_t)r * 2 + 1) * NE;
            k_scatter_add<<<Gs, B, 0, stream>>>(src, dst, hr, hnext);
        }
        if (l < NLAY - 1)
            k_cast_bf16<<<dim3((NN * NH / 4 + 255) / 256), B, 0, stream>>>(
                hnext, h_bf, NN * NH / 4, 1 /*relu*/);
    }

    // ---- global mean pool + MLP ----
    k_zero_f32<<<dim3((NG * NH + NG + 255) / 256), B, 0, stream>>>(sums, NG * NH + NG);
    k_pool_sum<<<dim3((NN * (NH / 4) + 255) / 256), B, 0, stream>>>(hnext, batch, sums);
    k_pool_cnt<<<dim3((NN + 255) / 256), B, 0, stream>>>(batch, cnt);
    k_mlp<<<dim3(NG), B, 0, stream>>>(sums, cnt, w1, b1, w2, b2, out);
}
